// conv_Update_51539607552581
// MI455X (gfx1250) — compile-verified, matrix operand order checked by reading, not run on hardware
//
#include <hip/hip_runtime.h>

// ---------------------------------------------------------------------------
// CDNA5 (gfx1250) fused modulated-deformable-conv block, bf16 WMMA pipeline
// with async global->LDS weight staging (ASYNCcnt) overlapping the gather.
// ---------------------------------------------------------------------------

typedef __attribute__((ext_vector_type(16))) __bf16 v16bf;
typedef __attribute__((ext_vector_type(2)))  __bf16 v2bf;
typedef __attribute__((ext_vector_type(8)))  float  v8f;
typedef __attribute__((ext_vector_type(4)))  unsigned int v4u;

union Frag16 { v4u u[2]; v16bf v; };

#define HWPIX 4096   // 64*64

__device__ __forceinline__ unsigned short f2bf(float f) {
  unsigned u = __float_as_uint(f);
  u += 0x7FFFu + ((u >> 16) & 1u);          // round-to-nearest-even
  return (unsigned short)(u >> 16);
}

// Pack two floats into one dword of bf16. Prefer the single-op packed
// hardware conversion when clang exposes it for gfx1250; otherwise manual RNE.
__device__ __forceinline__ unsigned pack2bf(float lo, float hi) {
#if __has_builtin(__builtin_amdgcn_cvt_pk_bf16_f32)
  union { v2bf v; unsigned u; } cv;
  cv.v = __builtin_amdgcn_cvt_pk_bf16_f32(lo, hi);
  return cv.u;
#else
  return (unsigned)f2bf(lo) | ((unsigned)f2bf(hi) << 16);
#endif
}

__device__ __forceinline__ int iclamp(int v, int lo, int hi) {
  return v < lo ? lo : (v > hi ? hi : v);
}

__device__ __forceinline__ v8f zero8() {
  v8f z = {0.f, 0.f, 0.f, 0.f, 0.f, 0.f, 0.f, 0.f};
  return z;
}

// Async global->LDS copies (tracked by ASYNCcnt). INST_OFFSET applies to both
// the LDS and global addresses (ISA 08_async_tensor §4.4), so one address pair
// serves consecutive chunks. Low 32 bits of a generic shared pointer == LDS
// byte address (aperture mapping, ISA §10.2).
__device__ __forceinline__ void async_copy64_b64(void* lds, const void* gsrc) {
  unsigned l = (unsigned)(uintptr_t)lds;
  asm volatile("global_load_async_to_lds_b64 %0, %1, off"
               :: "v"(l), "v"(gsrc) : "memory");
}

__device__ __forceinline__ void async_copy64B(void* lds, const void* gsrc) {
  unsigned l = (unsigned)(uintptr_t)lds;
  asm volatile(
      "global_load_async_to_lds_b128 %0, %1, off\n\t"
      "global_load_async_to_lds_b128 %0, %1, off offset:16\n\t"
      "global_load_async_to_lds_b128 %0, %1, off offset:32\n\t"
      "global_load_async_to_lds_b128 %0, %1, off offset:48"
      :: "v"(l), "v"(gsrc) : "memory");
}

__device__ __forceinline__ void wait_async0() {
  asm volatile("s_wait_asynccnt 0" ::: "memory");
}

// ---------------------------------------------------------------------------
// h = concat(x, msg1, msg2) along channels  -> [B=2][768][64][64] fp32
// ---------------------------------------------------------------------------
__global__ __launch_bounds__(256)
void concat_kernel(const float* __restrict__ x, const float* __restrict__ m1,
                   const float* __restrict__ m2, float* __restrict__ h, int n) {
  int i = blockIdx.x * 256 + threadIdx.x;
  if (i >= n) return;
  int b = i / (768 * HWPIX);
  int r = i - b * (768 * HWPIX);
  int c = r >> 12;
  int p = r & 4095;
  float v;
  if (c < 256)      v = x [((size_t)(b * 256 + c)        << 12) + p];
  else if (c < 512) v = m1[((size_t)(b * 256 + (c - 256)) << 12) + p];
  else              v = m2[((size_t)(b * 256 + (c - 512)) << 12) + p];
  h[i] = v;
}

// ---------------------------------------------------------------------------
// Pack fp32 weights [N][K] -> bf16 [Npad][K] (rows >= N zero-filled).
// ---------------------------------------------------------------------------
__global__ __launch_bounds__(256)
void pack_w(const float* __restrict__ src, unsigned short* __restrict__ dst,
            int N, int K, int total) {
  int i = blockIdx.x * 256 + threadIdx.x;
  if (i >= total) return;
  int n = i / K;
  int k = i - n * K;
  dst[i] = (n < N) ? f2bf(src[(size_t)n * K + k]) : (unsigned short)0;
}

// ---------------------------------------------------------------------------
// Regular 3x3 conv as im2col GEMM: om[M][32] = im2col(h)[M][K] * Wp[32][K]^T + b
// M-tile 128 (two image rows), N = 32, K-step 32. 8 waves, each 16x32.
// ---------------------------------------------------------------------------
__global__ __launch_bounds__(256)
void conv3x3_gemm(const float* __restrict__ h, const unsigned short* __restrict__ wp,
                  const float* __restrict__ bias, float* __restrict__ om, int Cin) {
  const int K = Cin * 9;
  __shared__ __align__(16) unsigned short As[128 * 32];  // 8 KB
  __shared__ __align__(16) unsigned short Bs[32 * 32];   // 2 KB

  const int tid   = threadIdx.x;
  const int m0    = blockIdx.x * 128;
  const int b     = m0 >> 12;
  const int ybase = (m0 >> 6) & 63;
  const float* hb = h + (size_t)b * Cin * HWPIX;

  const int wave = tid >> 5;
  const int lane = tid & 31;
  const int half = lane >> 4;
  const int l15  = lane & 15;
  const int wm   = wave * 16;

  // A-build geometry: this thread fills row r, 16 contiguous kk.
  const int ar   = tid >> 1;          // 0..127
  const int akk0 = (tid & 1) * 16;    // 0 or 16
  const int ax   = ar & 63;
  const int ay   = ybase + (ar >> 6);

  v8f acc0 = zero8();
  v8f acc1 = zero8();

  for (int k0 = 0; k0 < K; k0 += 32) {
    // --- B tile: 32x32 bf16, async copy (8B per thread) ---
    {
      int e = tid * 4;                 // 1024 elems, 4 per thread
      async_copy64_b64(Bs + e, wp + (size_t)(e >> 5) * K + k0 + (e & 31));
    }
    // --- A tile: 128x32 im2col bf16 (clamp+select, packed b128 stores) ---
    {
      int c = (k0 + akk0) / 9;
      int t = (k0 + akk0) - c * 9;
      float v[16];
      #pragma unroll
      for (int ee = 0; ee < 16; ++ee) {
        int t3 = t / 3;   // t in [0,9): cheap
        int yy = ay + t3 - 1;
        int xx = ax + (t - t3 * 3) - 1;
        bool ok = ((unsigned)yy < 64u) & ((unsigned)xx < 64u);
        int yyc = iclamp(yy, 0, 63), xxc = iclamp(xx, 0, 63);
        float val = hb[((size_t)c << 12) + (yyc << 6) + xxc];
        v[ee] = ok ? val : 0.f;
        if (++t == 9) { t = 0; ++c; }
      }
      v4u p0 = { pack2bf(v[0], v[1]),  pack2bf(v[2], v[3]),
                 pack2bf(v[4], v[5]),  pack2bf(v[6], v[7]) };
      v4u p1 = { pack2bf(v[8], v[9]),  pack2bf(v[10], v[11]),
                 pack2bf(v[12], v[13]), pack2bf(v[14], v[15]) };
      v4u* dst = (v4u*)(As + (ar << 5) + akk0);
      dst[0] = p0;
      dst[1] = p1;
    }
    wait_async0();
    __syncthreads();

    Frag16 a, b0, b1;
    const unsigned short* ap = As + ((wm + l15) << 5) + half * 8;
    a.u[0] = *(const v4u*)(ap);
    a.u[1] = *(const v4u*)(ap + 16);
    const unsigned short* bp0 = Bs + (l15 << 5) + half * 8;
    b0.u[0] = *(const v4u*)(bp0);
    b0.u[1] = *(const v4u*)(bp0 + 16);
    const unsigned short* bp1 = Bs + ((16 + l15) << 5) + half * 8;
    b1.u[0] = *(const v4u*)(bp1);
    b1.u[1] = *(const v4u*)(bp1 + 16);

    acc0 = __builtin_amdgcn_wmma_f32_16x16x32_bf16(false, a.v, false, b0.v,
                                                   (short)0, acc0, false, false);
    acc1 = __builtin_amdgcn_wmma_f32_16x16x32_bf16(false, a.v, false, b1.v,
                                                   (short)0, acc1, false, false);
    __syncthreads();
  }

  #pragma unroll
  for (int j = 0; j < 2; ++j) {
    int n = j * 16 + l15;
    float bz = (n < 27) ? bias[n] : 0.f;
    v8f acc = j ? acc1 : acc0;
    #pragma unroll
    for (int vv = 0; vv < 8; ++vv) {
      int m = m0 + wm + vv + half * 8;
      om[(size_t)m * 32 + n] = acc[vv] + bz;
    }
  }
}

// ---------------------------------------------------------------------------
// Deformable GEMM: D[m][n] = sum_k gather(h)[m][k] * Wp[n][k], fused BN+ReLU.
// M-tile 64 (one image row -> uniform b,y), N = 256, K-step 32.
// 8 waves as 4(M) x 2(N); each wave: 16 x 128 -> 8 accumulators.
// B tile staged with async LDS loads overlapping the bilinear A gather.
// finalMode: out = leakyrelu(x + gamma * relu(bn(acc))).
// ---------------------------------------------------------------------------
__global__ __launch_bounds__(256)
void deform_gemm(const float* __restrict__ hsrc, const float* __restrict__ om,
                 const unsigned short* __restrict__ wp,
                 const float* __restrict__ bng, const float* __restrict__ bnb,
                 const float* __restrict__ bnm, const float* __restrict__ bnv,
                 int Cin, int finalMode, const float* __restrict__ xres,
                 const float* __restrict__ gamma, float* __restrict__ dst) {
  const int K = Cin * 9;
  __shared__ __align__(16) unsigned short As[64 * 32];    // 4 KB
  __shared__ __align__(16) unsigned short Bs[256 * 32];   // 16 KB
  __shared__ float dwgt[576][4];                          // 9 KB
  __shared__ int   didx[576][4];                          // 9 KB

  const int tid = threadIdx.x;
  const int m0  = blockIdx.x * 64;
  const int b   = m0 >> 12;
  const int y   = (m0 >> 6) & 63;
  const float* hb = hsrc + (size_t)b * Cin * HWPIX;

  // --- per-(row, tap) bilinear descriptors ---
  for (int d = tid; d < 576; d += 256) {
    int r = d / 9;
    int t = d - r * 9;
    const float* omr = om + (size_t)(m0 + r) * 32;
    float dyv = omr[2 * t];
    float dxv = omr[2 * t + 1];
    float mk  = 1.f / (1.f + __expf(-omr[18 + t]));   // sigmoid(mask)
    int t3 = t / 3;
    float ys = dyv + (float)y + (float)(t3 - 1);
    float xs = dxv + (float)r + (float)(t - t3 * 3 - 1);
    float y0f = floorf(ys), x0f = floorf(xs);
    float wy = ys - y0f, wx = xs - x0f;
    int y0 = (int)y0f, x0 = (int)x0f;
    int y1 = y0 + 1, x1 = x0 + 1;
    float v00 = ((unsigned)y0 < 64u && (unsigned)x0 < 64u) ? 1.f : 0.f;
    float v01 = ((unsigned)y0 < 64u && (unsigned)x1 < 64u) ? 1.f : 0.f;
    float v10 = ((unsigned)y1 < 64u && (unsigned)x0 < 64u) ? 1.f : 0.f;
    float v11 = ((unsigned)y1 < 64u && (unsigned)x1 < 64u) ? 1.f : 0.f;
    int y0c = iclamp(y0, 0, 63), x0c = iclamp(x0, 0, 63);
    int y1c = iclamp(y1, 0, 63), x1c = iclamp(x1, 0, 63);
    dwgt[d][0] = (1.f - wy) * (1.f - wx) * mk * v00;
    dwgt[d][1] = (1.f - wy) * wx * mk * v01;
    dwgt[d][2] = wy * (1.f - wx) * mk * v10;
    dwgt[d][3] = wy * wx * mk * v11;
    didx[d][0] = (y0c << 6) + x0c;
    didx[d][1] = (y0c << 6) + x1c;
    didx[d][2] = (y1c << 6) + x0c;
    didx[d][3] = (y1c << 6) + x1c;
  }
  __syncthreads();

  const int wave = tid >> 5;
  const int lane = tid & 31;
  const int half = lane >> 4;
  const int l15  = lane & 15;
  const int wm   = (wave & 3) * 16;
  const int wn   = (wave >> 2) * 128;

  // A-build geometry: row ar, 8 contiguous kk starting at akk0.
  const int ar   = tid >> 2;          // 0..63
  const int akk0 = (tid & 3) * 8;

  v8f acc[8];
  #pragma unroll
  for (int j = 0; j < 8; ++j) acc[j] = zero8();

  for (int k0 = 0; k0 < K; k0 += 32) {
    // --- B tile: 256x32 bf16, async copy (64B per thread), overlaps gather ---
    async_copy64B(Bs + (tid << 5), wp + (size_t)tid * K + k0);

    // --- A tile: 64x32 bf16, gathered via precomputed descriptors ---
    {
      int c = (k0 + akk0) / 9;
      int t = (k0 + akk0) - c * 9;
      float v[8];
      #pragma unroll
      for (int ee = 0; ee < 8; ++ee) {
        int d = ar * 9 + t;
        const float* p = hb + ((size_t)c << 12);
        v[ee] = dwgt[d][0] * p[didx[d][0]] + dwgt[d][1] * p[didx[d][1]]
              + dwgt[d][2] * p[didx[d][2]] + dwgt[d][3] * p[didx[d][3]];
        if (++t == 9) { t = 0; ++c; }
      }
      v4u pk = { pack2bf(v[0], v[1]), pack2bf(v[2], v[3]),
                 pack2bf(v[4], v[5]), pack2bf(v[6], v[7]) };
      *(v4u*)(As + (ar << 5) + akk0) = pk;
    }
    wait_async0();
    __syncthreads();

    Frag16 a;
    const unsigned short* ap = As + ((wm + l15) << 5) + half * 8;
    a.u[0] = *(const v4u*)(ap);
    a.u[1] = *(const v4u*)(ap + 16);
    #pragma unroll
    for (int j = 0; j < 8; ++j) {
      Frag16 bf;
      const unsigned short* bp = Bs + ((wn + j * 16 + l15) << 5) + half * 8;
      bf.u[0] = *(const v4u*)(bp);
      bf.u[1] = *(const v4u*)(bp + 16);
      acc[j] = __builtin_amdgcn_wmma_f32_16x16x32_bf16(false, a.v, false, bf.v,
                                                       (short)0, acc[j], false, false);
    }
    __syncthreads();
  }

  // --- epilogue: BN + ReLU (+ residual / gamma / leaky-ReLU) ---
  // Per lane the 8 accumulator elements are 8 consecutive x positions for a
  // fixed channel n -> two float4 stores per fragment (coalesced 32B runs).
  float gam = finalMode ? gamma[0] : 0.f;
  #pragma unroll
  for (int j = 0; j < 8; ++j) {
    int n = wn + j * 16 + l15;
    float sc = bng[n] * rsqrtf(bnv[n] + 1e-5f);
    float sh = bnb[n] - bnm[n] * sc;
    size_t idx0 = (((size_t)(b * 256 + n)) << 12) + (y << 6) + wm + half * 8;
    float r[8];
    #pragma unroll
    for (int vv = 0; vv < 8; ++vv)
      r[vv] = fmaxf(acc[j][vv] * sc + sh, 0.f);
    if (finalMode) {
      float4 x0 = *(const float4*)(xres + idx0);
      float4 x1 = *(const float4*)(xres + idx0 + 4);
      float o[8] = { x0.x + gam * r[0], x0.y + gam * r[1],
                     x0.z + gam * r[2], x0.w + gam * r[3],
                     x1.x + gam * r[4], x1.y + gam * r[5],
                     x1.z + gam * r[6], x1.w + gam * r[7] };
      #pragma unroll
      for (int vv = 0; vv < 8; ++vv)
        o[vv] = (o[vv] >= 0.f) ? o[vv] : 0.01f * o[vv];
      float4 s0 = { o[0], o[1], o[2], o[3] };
      float4 s1 = { o[4], o[5], o[6], o[7] };
      *(float4*)(dst + idx0)     = s0;
      *(float4*)(dst + idx0 + 4) = s1;
    } else {
      float4 s0 = { r[0], r[1], r[2], r[3] };
      float4 s1 = { r[4], r[5], r[6], r[7] };
      *(float4*)(dst + idx0)     = s0;
      *(float4*)(dst + idx0 + 4) = s1;
    }
  }
}

// ---------------------------------------------------------------------------
// Host orchestration (all on `stream`, graph-capture safe).
// ---------------------------------------------------------------------------
extern "C" void kernel_launch(void* const* d_in, const int* in_sizes, int n_in,
                              void* d_out, int out_size, void* d_ws, size_t ws_size,
                              hipStream_t stream) {
  (void)in_sizes; (void)n_in; (void)out_size; (void)ws_size;

  const float* x      = (const float*)d_in[0];
  const float* msg1   = (const float*)d_in[1];
  const float* msg2   = (const float*)d_in[2];
  const float* w_off1 = (const float*)d_in[3];
  const float* b_off1 = (const float*)d_in[4];
  const float* w1     = (const float*)d_in[5];
  const float* bn1g   = (const float*)d_in[6];
  const float* bn1b   = (const float*)d_in[7];
  const float* bn1m   = (const float*)d_in[8];
  const float* bn1v   = (const float*)d_in[9];
  const float* w_off2 = (const float*)d_in[10];
  const float* b_off2 = (const float*)d_in[11];
  const float* w2     = (const float*)d_in[12];
  const float* bn2g   = (const float*)d_in[13];
  const float* bn2b   = (const float*)d_in[14];
  const float* bn2m   = (const float*)d_in[15];
  const float* bn2v   = (const float*)d_in[16];
  const float* gamma  = (const float*)d_in[17];

  char* ws = (char*)d_ws;
  auto carve = [&](size_t bytes) -> char* {
    char* p = ws;
    ws += (bytes + 255) & ~(size_t)255;
    return p;
  };
  float*          h    = (float*)carve((size_t)2 * 768 * HWPIX * 4);  // 25.2 MB
  float*          h1   = (float*)carve((size_t)2 * 256 * HWPIX * 4);  //  8.4 MB
  float*          om   = (float*)carve((size_t)8192 * 32 * 4);        //  1.0 MB
  unsigned short* wpo1 = (unsigned short*)carve((size_t)32  * 6912 * 2);
  unsigned short* wpm1 = (unsigned short*)carve((size_t)256 * 6912 * 2);
  unsigned short* wpo2 = (unsigned short*)carve((size_t)32  * 2304 * 2);
  unsigned short* wpm2 = (unsigned short*)carve((size_t)256 * 2304 * 2);

  // prep
  concat_kernel<<<(2 * 768 * HWPIX + 255) / 256, 256, 0, stream>>>(
      x, msg1, msg2, h, 2 * 768 * HWPIX);
  pack_w<<<(32  * 6912 + 255) / 256, 256, 0, stream>>>(w_off1, wpo1, 27, 6912, 32  * 6912);
  pack_w<<<(256 * 6912 + 255) / 256, 256, 0, stream>>>(w1,     wpm1, 256, 6912, 256 * 6912);
  pack_w<<<(32  * 2304 + 255) / 256, 256, 0, stream>>>(w_off2, wpo2, 27, 2304, 32  * 2304);
  pack_w<<<(256 * 2304 + 255) / 256, 256, 0, stream>>>(w2,     wpm2, 256, 2304, 256 * 2304);

  // block 1: offsets, deform GEMM + BN/ReLU -> h1
  conv3x3_gemm<<<64, 256, 0, stream>>>(h, wpo1, b_off1, om, 768);
  deform_gemm<<<128, 256, 0, stream>>>(h, om, wpm1, bn1g, bn1b, bn1m, bn1v,
                                       768, 0, x, gamma, h1);

  // block 2: offsets, deform GEMM + BN/ReLU + residual/leaky -> d_out
  conv3x3_gemm<<<64, 256, 0, stream>>>(h1, wpo2, b_off2, om, 256);
  deform_gemm<<<128, 256, 0, stream>>>(h1, om, wpm2, bn2g, bn2b, bn2m, bn2v,
                                       256, 1, x, gamma, (float*)d_out);
}
